// Decoder_55963423867149
// MI455X (gfx1250) — matrix operand order repaired and checked by reading, not compiled, not run on hardware
//
#include <hip/hip_runtime.h>
#include <math.h>

typedef _Float16 h16_t;
typedef __attribute__((ext_vector_type(16))) _Float16 v16h;
typedef __attribute__((ext_vector_type(8)))  _Float16 v8h;
typedef __attribute__((ext_vector_type(8)))  float    v8f;

static constexpr int T_STEPS = 48;
static constexpr int B_SZ    = 64;
static constexpr int L_SZ    = 100;
static constexpr int DECD    = 512;   // DEC == ENC == DW == ATT == 512
static constexpr int HID3    = 3 * DECD;

// ---------------------------------------------------------------------------
// CDNA5 async global->LDS copy (ASYNCcnt-tracked), via inline asm
// vdst = VGPR holding wave-relative LDS byte offset; vaddr = 64-bit global.
// ---------------------------------------------------------------------------
__device__ __forceinline__ void async_ld_b128(uint32_t lds_off, const void* gptr) {
    asm volatile("global_load_async_to_lds_b128 %0, %1, off"
                 :: "v"(lds_off), "v"((unsigned long long)(uintptr_t)gptr)
                 : "memory");
}
__device__ __forceinline__ void wait_async0() {
    asm volatile("s_wait_asynccnt 0x0" ::: "memory");
}

// ---------------------------------------------------------------------------
// WMMA fragment helpers (layouts per CDNA5 ISA 7.12.2, wave32)
// ---------------------------------------------------------------------------
__device__ __forceinline__ v16h pack16(v8h lo, v8h hi) {
    union { v16h v; v8h h[2]; } u;
    u.h[0] = lo; u.h[1] = hi;
    return u.v;
}

// A (16x32 f16, row-major source, lda in halves). lanes 0-15: K kh*8..+7 then
// 16+kh*8..+7 ; lanes 16-31 shift K base by 8.
__device__ __forceinline__ v16h load_a_frag(const h16_t* __restrict__ X, int lda,
                                            int k0, int lane) {
    const int m  = lane & 15;
    const int kh = (lane >> 4) & 1;
    const h16_t* p = X + (size_t)m * lda + k0 + kh * 8;
    v8h lo = *(const v8h*)(p);
    v8h hi = *(const v8h*)(p + 16);
    return pack16(lo, hi);
}

// B (32x16 f16) where B = W^T and W is [N][K] row-major: lane column n reads
// K kh*8..+7 then 16+kh*8..+7 of W's row (n0+n).  Works for global or LDS source.
__device__ __forceinline__ v16h load_b_frag(const h16_t* W, int ldw,
                                            int n0, int k0, int lane) {
    const int n  = lane & 15;
    const int kh = (lane >> 4) & 1;
    const h16_t* p = W + (size_t)(n0 + n) * ldw + k0 + kh * 8;
    v8h lo = *(const v8h*)(p);
    v8h hi = *(const v8h*)(p + 16);
    return pack16(lo, hi);
}

__device__ __forceinline__ v8f wmma_f16(v16h a, v16h b, v8f c) {
    return __builtin_amdgcn_wmma_f32_16x16x32_f16(false, a, false, b,
                                                  (short)0, c, false, false);
}

__device__ __forceinline__ float sigmoidf_(float x) {
    return 1.0f / (1.0f + expf(-x));
}

// ---------------------------------------------------------------------------
// Precompute kernels
// ---------------------------------------------------------------------------
__global__ void cvt_f32_to_f16(const float* __restrict__ in,
                               h16_t* __restrict__ out, int n) {
    int i = blockIdx.x * blockDim.x + threadIdx.x;
    if (i < n) out[i] = (h16_t)in[i];
}

__global__ void init_h_kernel(const float* __restrict__ hidden,
                              float* __restrict__ h32, h16_t* __restrict__ h16,
                              int n) {
    int i = blockIdx.x * blockDim.x + threadIdx.x;
    if (i < n) { float v = hidden[i]; h32[i] = v; h16[i] = (h16_t)v; }
}

// emb16[t][b][e] = word_lut[input_ids[t][b]][e]
__global__ void emb_gather_kernel(const int* __restrict__ ids,
                                  const float* __restrict__ lut,
                                  h16_t* __restrict__ emb16) {
    int i = blockIdx.x * blockDim.x + threadIdx.x;
    if (i < T_STEPS * B_SZ * DECD) {
        int e  = i & (DECD - 1);
        int tb = i >> 9;
        int id = ids[tb];
        emb16[i] = (h16_t)lut[(size_t)id * DECD + e];
    }
}

// ctx16[b][l][e] = context[l][b][e]  (f16, GEMM-ready row-major over (b,l))
__global__ void ctx_cvt_kernel(const float* __restrict__ context,
                               h16_t* __restrict__ ctx16) {
    int i = blockIdx.x * blockDim.x + threadIdx.x;
    if (i < B_SZ * L_SZ * DECD) {
        int e  = i & (DECD - 1);
        int bl = i >> 9;
        int l  = bl % L_SZ;
        int b  = bl / L_SZ;
        ctx16[i] = (h16_t)context[((size_t)l * B_SZ + b) * DECD + e];
    }
}

// ---------------------------------------------------------------------------
// Generic Y[M][N] = A[M][K] @ W[N][K]^T (+bias), WMMA f16->f32
// grid: (N/16, M/64) ; block: 128 (4 waves, one 16-row M-tile per wave)
// ---------------------------------------------------------------------------
__global__ void gemm_xwT_kernel(const h16_t* __restrict__ A, int lda,
                                const h16_t* __restrict__ W, int ldw,
                                const float* __restrict__ bias,
                                float* __restrict__ Y, int ldy, int K) {
    const int lane = threadIdx.x & 31;
    const int wave = threadIdx.x >> 5;
    const int m0   = blockIdx.y * 64 + wave * 16;
    const int n0   = blockIdx.x * 16;
    v8f acc = {};
    const h16_t* Arow = A + (size_t)m0 * lda;
    for (int k0 = 0; k0 < K; k0 += 32) {
        v16h a = load_a_frag(Arow, lda, k0, lane);
        v16h b = load_b_frag(W, ldw, n0, k0, lane);
        acc = wmma_f16(a, b, acc);
    }
    const int n  = n0 + (lane & 15);
    const float bv = bias ? bias[n] : 0.0f;
    const int mb = m0 + ((lane >> 4) << 3);
#pragma unroll
    for (int v = 0; v < 8; ++v)
        Y[(size_t)(mb + v) * ldy + n] = acc[v] + bv;
}

// ---------------------------------------------------------------------------
// Fused GRU step with async-staged weight panels in LDS.
// Block n0 needs W_ih rows {n0, n0+512, n0+1024}+0..15 (48 x 1024 halves)
// and W_hh rows likewise (48 x 512 halves) -> 144 KB dynamic LDS.
// grid: 32 blocks x 128 threads.
// ---------------------------------------------------------------------------
__global__ void gru_step_kernel(const h16_t* __restrict__ emb_t,
                                const h16_t* __restrict__ ctxcur16,
                                const h16_t* __restrict__ hprev16,
                                const float* __restrict__ hprev32,
                                const h16_t* __restrict__ Wih16,   // [1536][1024]
                                const h16_t* __restrict__ Whh16,   // [1536][512]
                                const float* __restrict__ b_ih,
                                const float* __restrict__ b_hh,
                                float* __restrict__ hnew32,
                                h16_t* __restrict__ hnew16,
                                float* __restrict__ ghid_out,      // g_hiddens[t]
                                float* __restrict__ hfin_out) {    // h_fin (last wins)
    extern __shared__ __align__(16) h16_t smem[];
    h16_t* sWih = smem;                 // [48][1024]
    h16_t* sWhh = smem + 48 * 1024;     // [48][512]

    const int lane = threadIdx.x & 31;
    const int wave = threadIdx.x >> 5;
    const int m0   = wave * 16;
    const int n0   = blockIdx.x * 16;

    // ---- stage weight panels: global -> LDS via CDNA5 async copies ----
    for (int t = threadIdx.x; t < 48 * 128; t += 128) {   // 128 b128 chunks / row
        int r = t >> 7;
        int c = (t & 127) * 8;
        int g = (r < 16) ? (n0 + r) : (r < 32) ? (512 + n0 + r - 16)
                                               : (1024 + n0 + r - 32);
        async_ld_b128((uint32_t)(uintptr_t)&sWih[r * 1024 + c],
                      Wih16 + (size_t)g * 1024 + c);
    }
    for (int t = threadIdx.x; t < 48 * 64; t += 128) {    // 64 b128 chunks / row
        int r = t >> 6;
        int c = (t & 63) * 8;
        int g = (r < 16) ? (n0 + r) : (r < 32) ? (512 + n0 + r - 16)
                                               : (1024 + n0 + r - 32);
        async_ld_b128((uint32_t)(uintptr_t)&sWhh[r * 512 + c],
                      Whh16 + (size_t)g * 512 + c);
    }
    wait_async0();
    __syncthreads();

    v8f cir = {}, ciz = {}, cin = {}, chr2 = {}, chz = {}, chn = {};

    for (int k0 = 0; k0 < 1024; k0 += 32) {           // x = [emb | ctx]
        const h16_t* base = (k0 < 512) ? emb_t : ctxcur16;
        const int kk = k0 & 511;
        v16h a  = load_a_frag(base + (size_t)m0 * DECD, DECD, kk, lane);
        v16h br = load_b_frag(sWih, 1024, 0,  k0, lane);
        v16h bz = load_b_frag(sWih, 1024, 16, k0, lane);
        v16h bn = load_b_frag(sWih, 1024, 32, k0, lane);
        cir = wmma_f16(a, br, cir);
        ciz = wmma_f16(a, bz, ciz);
        cin = wmma_f16(a, bn, cin);
    }
    for (int k0 = 0; k0 < 512; k0 += 32) {            // h @ W_hh^T
        v16h a  = load_a_frag(hprev16 + (size_t)m0 * DECD, DECD, k0, lane);
        v16h br = load_b_frag(sWhh, 512, 0,  k0, lane);
        v16h bz = load_b_frag(sWhh, 512, 16, k0, lane);
        v16h bn = load_b_frag(sWhh, 512, 32, k0, lane);
        chr2 = wmma_f16(a, br, chr2);
        chz  = wmma_f16(a, bz, chz);
        chn  = wmma_f16(a, bn, chn);
    }
    const int n   = n0 + (lane & 15);
    const float bir = b_ih[n], biz = b_ih[n + 512], bin = b_ih[n + 1024];
    const float bhr = b_hh[n], bhz = b_hh[n + 512], bhn = b_hh[n + 1024];
    const int mb = m0 + ((lane >> 4) << 3);
#pragma unroll
    for (int v = 0; v < 8; ++v) {
        const int m = mb + v;
        float r  = sigmoidf_(cir[v] + bir + chr2[v] + bhr);
        float z  = sigmoidf_(ciz[v] + biz + chz[v] + bhz);
        float nn = tanhf(cin[v] + bin + r * (chn[v] + bhn));
        float h1 = (1.0f - z) * nn + z * hprev32[(size_t)m * DECD + n];
        hnew32[(size_t)m * DECD + n] = h1;
        hnew16[(size_t)m * DECD + n] = (h16_t)h1;
        ghid_out[(size_t)m * DECD + n] = h1;
        hfin_out[(size_t)m * DECD + n] = h1;
    }
}

// ---------------------------------------------------------------------------
// Attention: energy = tanh(pre + q) . Wv + mask ; softmax over L ; wctx.
// grid: B blocks x 128 threads.
// ---------------------------------------------------------------------------
__global__ void attn_kernel(const float* __restrict__ q,        // [B][512]
                            const float* __restrict__ pre,      // [B][L][512]
                            const float* __restrict__ Wv,       // [512]
                            const float* __restrict__ mask,     // [B][L]
                            const float* __restrict__ context,  // [L][B][512]
                            float* __restrict__ score_out,      // c_outputs[t]
                            float* __restrict__ attn_last,      // last wins
                            float* __restrict__ wctx32,
                            h16_t* __restrict__ wctx16,
                            float* __restrict__ ctx_fin) {      // last wins
    const int b    = blockIdx.x;
    const int tid  = threadIdx.x;
    const int wave = tid >> 5;
    const int lane = tid & 31;
    __shared__ float sE[L_SZ];
    __shared__ float sS[L_SZ];

    for (int l = wave; l < L_SZ; l += 4) {
        const float* pr = pre + ((size_t)b * L_SZ + l) * DECD;
        const float* qb = q + (size_t)b * DECD;
        float s = 0.0f;
        for (int a = lane; a < DECD; a += 32)
            s += tanhf(pr[a] + qb[a]) * Wv[a];
        for (int off = 16; off > 0; off >>= 1)
            s += __shfl_xor(s, off, 32);
        if (lane == 0) {
            float mb = (mask[(size_t)b * L_SZ + l] > 0.0f) ? -1e6f : 0.0f;
            sE[l] = s + mb;
        }
    }
    __syncthreads();
    if (tid == 0) {
        float mx = -3.4e38f;
        for (int l = 0; l < L_SZ; ++l) mx = fmaxf(mx, sE[l]);
        float sum = 0.0f;
        for (int l = 0; l < L_SZ; ++l) { float e = expf(sE[l] - mx); sS[l] = e; sum += e; }
        float inv = 1.0f / sum;
        for (int l = 0; l < L_SZ; ++l) sS[l] *= inv;
    }
    __syncthreads();
    for (int l = tid; l < L_SZ; l += blockDim.x) {
        score_out[(size_t)b * L_SZ + l] = sS[l];
        attn_last[(size_t)b * L_SZ + l] = sS[l];
    }
    for (int e = tid; e < DECD; e += blockDim.x) {
        float acc = 0.0f;
        for (int l = 0; l < L_SZ; ++l)
            acc += sS[l] * context[((size_t)l * B_SZ + b) * DECD + e];
        wctx32[(size_t)b * DECD + e] = acc;
        wctx16[(size_t)b * DECD + e] = (h16_t)acc;
        ctx_fin[(size_t)b * DECD + e] = acc;
    }
}

// copy_p = sigmoid([h1|wctx] @ W_copy^T + b_copy)  — 64 threads, 1 block
__global__ void copy_gate_kernel(const float* __restrict__ h1,
                                 const float* __restrict__ wctx,
                                 const float* __restrict__ W_copy, // [1024]
                                 const float* __restrict__ b_copy,
                                 float* __restrict__ out) {        // copy_gates[t]
    int b = threadIdx.x;
    float s = b_copy[0];
    for (int j = 0; j < DECD; ++j) s += h1[(size_t)b * DECD + j] * W_copy[j];
    for (int j = 0; j < DECD; ++j) s += wctx[(size_t)b * DECD + j] * W_copy[DECD + j];
    out[b] = 1.0f / (1.0f + expf(-s));
}

// readout + maxout(pool=2): ro = [emb|h1|wctx] @ W_read^T + b_read
// Weight panel (16 rows x 1536 halves = 48 KB) async-staged in LDS.
// grid: 32 blocks x 128 threads
__global__ void readout_kernel(const h16_t* __restrict__ emb_t,
                               const h16_t* __restrict__ h1_16,
                               const h16_t* __restrict__ wctx16,
                               const h16_t* __restrict__ Wread16, // [512][1536]
                               const float* __restrict__ b_read,
                               float* __restrict__ gout) {        // g_outputs[t] [64][256]
    extern __shared__ __align__(16) h16_t sW[];                   // [16][1536]
    const int lane = threadIdx.x & 31;
    const int wave = threadIdx.x >> 5;
    const int m0   = wave * 16;
    const int n0   = blockIdx.x * 16;

    for (int t = threadIdx.x; t < 16 * 192; t += 128) {   // 192 b128 chunks / row
        int r = t / 192;
        int c = (t % 192) * 8;
        async_ld_b128((uint32_t)(uintptr_t)&sW[r * 1536 + c],
                      Wread16 + (size_t)(n0 + r) * 1536 + c);
    }
    wait_async0();
    __syncthreads();

    v8f acc = {};
    for (int k0 = 0; k0 < 1536; k0 += 32) {
        const h16_t* base = (k0 < 512) ? emb_t : ((k0 < 1024) ? h1_16 : wctx16);
        const int kk = k0 & 511;
        v16h a = load_a_frag(base + (size_t)m0 * DECD, DECD, kk, lane);
        v16h b = load_b_frag(sW, 1536, 0, k0, lane);
        acc = wmma_f16(a, b, acc);
    }
    const int n  = n0 + (lane & 15);
    const float bv = b_read[n];
    const int mb = m0 + ((lane >> 4) << 3);
#pragma unroll
    for (int v = 0; v < 8; ++v) {
        float ro    = acc[v] + bv;
        float other = __shfl_xor(ro, 1, 32);      // partner column n^1
        if ((lane & 1) == 0) {
            float mo = fmaxf(ro, other);
            gout[(size_t)(mb + v) * 256 + (n >> 1)] = mo;
        }
    }
}

// ---------------------------------------------------------------------------
// Host launcher
// ---------------------------------------------------------------------------
extern "C" void kernel_launch(void* const* d_in, const int* in_sizes, int n_in,
                              void* d_out, int out_size, void* d_ws, size_t ws_size,
                              hipStream_t stream) {
    (void)in_sizes; (void)n_in; (void)out_size;

    const int*   input_ids = (const int*)  d_in[1];
    const float* hidden    = (const float*)d_in[2];
    const float* context   = (const float*)d_in[3];
    const float* mask      = (const float*)d_in[4];
    const float* init_att  = (const float*)d_in[5];
    const float* word_lut  = (const float*)d_in[8];
    const float* W_ih      = (const float*)d_in[9];
    const float* W_hh      = (const float*)d_in[10];
    const float* b_ih      = (const float*)d_in[11];
    const float* b_hh      = (const float*)d_in[12];
    const float* W_pre     = (const float*)d_in[13];
    const float* b_pre     = (const float*)d_in[14];
    const float* W_q       = (const float*)d_in[15];
    const float* W_v       = (const float*)d_in[16];
    const float* W_copy    = (const float*)d_in[17];
    const float* b_copy    = (const float*)d_in[18];
    const float* W_read    = (const float*)d_in[19];
    const float* b_read    = (const float*)d_in[20];

    // --- output regions (reference tuple order, flat fp32) ---
    float* out          = (float*)d_out;
    float* out_g        = out;                               // [48][64][256]
    float* out_c        = out + 786432;                      // [48][64][100]
    float* out_copy     = out + 786432 + 307200;             // [48][64]
    float* out_hfin     = out + 786432 + 307200 + 3072;      // [1][64][512]
    float* out_attnlast = out_hfin + 32768;                  // [64][100]
    float* out_ctxfin   = out_attnlast + 6400;               // [64][512]
    float* out_ghid     = out_ctxfin + 32768;                // [48][1][64][512]

    // --- workspace carve (bump allocator, 256B aligned) ---
    char* ws = (char*)d_ws;
    size_t off = 0;
    auto alloc = [&](size_t bytes) -> void* {
        void* p = ws + off;
        off = (off + bytes + 255) & ~(size_t)255;
        return p;
    };
    h16_t* Wih16   = (h16_t*)alloc((size_t)HID3 * 1024 * 2);
    h16_t* Whh16   = (h16_t*)alloc((size_t)HID3 * 512 * 2);
    h16_t* Wq16    = (h16_t*)alloc((size_t)512 * 512 * 2);
    h16_t* Wread16 = (h16_t*)alloc((size_t)512 * 1536 * 2);
    h16_t* Wpre16  = (h16_t*)alloc((size_t)512 * 512 * 2);
    h16_t* emb16   = (h16_t*)alloc((size_t)T_STEPS * B_SZ * DECD * 2);
    h16_t* ctx16   = (h16_t*)alloc((size_t)B_SZ * L_SZ * DECD * 2);
    float* pre32   = (float*)alloc((size_t)B_SZ * L_SZ * DECD * 4);
    float* q32     = (float*)alloc((size_t)B_SZ * DECD * 4);
    float* h32a    = (float*)alloc((size_t)B_SZ * DECD * 4);
    float* h32b    = (float*)alloc((size_t)B_SZ * DECD * 4);
    h16_t* h16a    = (h16_t*)alloc((size_t)B_SZ * DECD * 2);
    h16_t* h16b    = (h16_t*)alloc((size_t)B_SZ * DECD * 2);
    float* wctx32  = (float*)alloc((size_t)B_SZ * DECD * 4);
    h16_t* wctx16  = (h16_t*)alloc((size_t)B_SZ * DECD * 2);
    if (off > ws_size) return;   // workspace too small; bail deterministically

    float* h32[2] = { h32a, h32b };
    h16_t* h16[2] = { h16a, h16b };

    auto cvt = [&](const float* src, h16_t* dst, int n) {
        cvt_f32_to_f16<<<(n + 255) / 256, 256, 0, stream>>>(src, dst, n);
    };

    // ---- one-time precompute ----
    cvt(W_ih,   Wih16,   HID3 * 1024);
    cvt(W_hh,   Whh16,   HID3 * 512);
    cvt(W_q,    Wq16,    512 * 512);
    cvt(W_read, Wread16, 512 * 1536);
    cvt(W_pre,  Wpre16,  512 * 512);
    cvt(init_att, wctx16, B_SZ * DECD);

    {
        int n = T_STEPS * B_SZ * DECD;
        emb_gather_kernel<<<(n + 255) / 256, 256, 0, stream>>>(input_ids, word_lut, emb16);
    }
    {
        int n = B_SZ * L_SZ * DECD;
        ctx_cvt_kernel<<<(n + 255) / 256, 256, 0, stream>>>(context, ctx16);
    }
    init_h_kernel<<<(B_SZ * DECD + 255) / 256, 256, 0, stream>>>(hidden, h32[0], h16[0],
                                                                 B_SZ * DECD);
    // pre = ctx @ W_pre^T + b_pre  :  M=6400, N=512, K=512
    gemm_xwT_kernel<<<dim3(32, 100), 128, 0, stream>>>(ctx16, DECD, Wpre16, DECD,
                                                       b_pre, pre32, DECD, DECD);

    const size_t gruLds  = (size_t)(48 * 1024 + 48 * 512) * sizeof(h16_t); // 144 KB
    const size_t readLds = (size_t)(16 * 1536) * sizeof(h16_t);            // 48 KB

    // ---- recurrent steps ----
    for (int t = 0; t < T_STEPS; ++t) {
        int rd = t & 1, wr = rd ^ 1;
        const h16_t* emb_t = emb16 + (size_t)t * B_SZ * DECD;

        gru_step_kernel<<<32, 128, gruLds, stream>>>(
            emb_t, wctx16, h16[rd], h32[rd], Wih16, Whh16, b_ih, b_hh,
            h32[wr], h16[wr], out_ghid + (size_t)t * B_SZ * DECD, out_hfin);

        gemm_xwT_kernel<<<dim3(32, 1), 128, 0, stream>>>(
            h16[wr], DECD, Wq16, DECD, nullptr, q32, DECD, DECD);

        attn_kernel<<<B_SZ, 128, 0, stream>>>(
            q32, pre32, W_v, mask, context,
            out_c + (size_t)t * B_SZ * L_SZ, out_attnlast,
            wctx32, wctx16, out_ctxfin);

        copy_gate_kernel<<<1, 64, 0, stream>>>(
            h32[wr], wctx32, W_copy, b_copy, out_copy + (size_t)t * B_SZ);

        readout_kernel<<<32, 128, readLds, stream>>>(
            emb_t, h16[wr], wctx16, Wread16, b_read,
            out_g + (size_t)t * B_SZ * 256);
    }
}
